// FractalGraphEncoder_54846732370025
// MI455X (gfx1250) — matrix operand order repaired and checked by reading, not compiled, not run on hardware
//
#include <hip/hip_runtime.h>

#define N_NODES 50000
#define N_EDGES 800000
#define N_GRAPHS 8
#define D_IN 384
#define D_HID 512
#define D_OUT 2560

typedef __attribute__((ext_vector_type(16))) __bf16 v16bf;
typedef __attribute__((ext_vector_type(8)))  float  v8f;

// ---------------- GEMM tiling ----------------
#define BM 128
#define BN 64
#define BK 32
#define LDAH 40   // padded bf16 stride (80B: 16B aligned, conflict-free)

// -------- CDNA5 async global->LDS (ASYNCcnt-tracked, 16B per lane) --------
__device__ __forceinline__ void async_ld_b128(const void* gptr, void* lptr)
{
    unsigned int lds = (unsigned int)(size_t)lptr;            // LDS offset (flat low bits)
    unsigned long long ga = (unsigned long long)(size_t)gptr; // 64-bit global address
    asm volatile("global_load_async_to_lds_b128 %0, %1, off"
                 :: "v"(lds), "v"(ga) : "memory");
}

// C = relu( A @ B^T_planes + bias ). A given as bf16 hi/lo planes, row-major [M,K].
// B given as bf16 hi/lo planes of B-transposed, row-major [N,K].
// bf16x3 emulated fp32: hi*hi + hi*lo + lo*hi, f32 accumulate.
// poolMode: accumulate per-graph column sums instead of storing C.
__global__ void __launch_bounds__(256)
wmma_gemm_bf16(const __bf16* __restrict__ Ahi, const __bf16* __restrict__ Alo,
               const __bf16* __restrict__ BThi, const __bf16* __restrict__ BTlo,
               const float* __restrict__ bias,
               float* __restrict__ Cout, float* __restrict__ poolSums,
               const int* __restrict__ batch,
               int M, int N, int K, int poolMode)
{
    __shared__ __align__(16) __bf16 sA[2][2][BM][LDAH]; // [buf][hi/lo]
    __shared__ __align__(16) __bf16 sB[2][2][BN][LDAH];

    const int tid  = threadIdx.x;
    const int lane = tid & 31;
    const int wid  = tid >> 5;
    const int wm   = wid & 3;          // 4 wave-rows of 32
    const int wn   = wid >> 2;         // 2 wave-cols of 32
    const int lh   = lane & 15;
    const bool hiHalf = lane >= 16;

    const int mBase = blockIdx.y * BM;
    const int nBase = blockIdx.x * BN;

    int fastGraph = -1;
    if (poolMode) {
        if (mBase + BM <= M) {
            int g0 = batch[mBase];
            int g1 = batch[mBase + BM - 1];
            if (g0 == g1) fastGraph = g0;
        }
    }

    v8f acc[2][2];
#pragma unroll
    for (int i = 0; i < 2; ++i)
#pragma unroll
        for (int j = 0; j < 2; ++j)
#pragma unroll
            for (int e = 0; e < 8; ++e) acc[i][j][e] = 0.0f;

    // per-thread async chunk coordinates (6 x b128 per thread per tile)
    const int ar0 = (tid) >> 1,        as0 = ((tid) & 1) << 3;       // A chunk 0
    const int ar1 = (tid + 256) >> 1,  as1 = ((tid + 256) & 1) << 3; // A chunk 1
    const int br  = tid >> 2,          bs  = (tid & 3) << 3;         // B chunk
    int agr0 = mBase + ar0; if (agr0 > M - 1) agr0 = M - 1;
    int agr1 = mBase + ar1; if (agr1 > M - 1) agr1 = M - 1;
    const int bgr = nBase + br;        // N is a multiple of BN

    auto issueTile = [&](int buf, int k0) {
        size_t goA0 = (size_t)agr0 * K + k0 + as0;
        size_t goA1 = (size_t)agr1 * K + k0 + as1;
        size_t goB  = (size_t)bgr  * K + k0 + bs;
        async_ld_b128(Ahi + goA0, &sA[buf][0][ar0][as0]);
        async_ld_b128(Alo + goA0, &sA[buf][1][ar0][as0]);
        async_ld_b128(Ahi + goA1, &sA[buf][0][ar1][as1]);
        async_ld_b128(Alo + goA1, &sA[buf][1][ar1][as1]);
        async_ld_b128(BThi + goB, &sB[buf][0][br][bs]);
        async_ld_b128(BTlo + goB, &sB[buf][1][br][bs]);
    };

    const int kbA = hiHalf ? 8 : 0;    // A frag: K = kb+e (e<8), kb+16+(e-8)
    const int kbB = hiHalf ? 8 : 0;    // B frag: K = kb+e (e<8), kb+16+(e-8)
    const int T = K / BK;

    union U { uint4 q[2]; v16bf v; };

    issueTile(0, 0);
    for (int t = 0; t < T; ++t) {
        if (t + 1 < T) {
            issueTile((t + 1) & 1, (t + 1) * BK);
            asm volatile("s_wait_asynccnt 0x6" ::: "memory"); // prev tile landed
        } else {
            asm volatile("s_wait_asynccnt 0x0" ::: "memory");
        }
        __syncthreads();

        const int buf = t & 1;
        U aHi[2], aLo[2], bHi[2], bLo[2];
#pragma unroll
        for (int ms = 0; ms < 2; ++ms) {
            int m = wm * 32 + ms * 16 + lh;
            aHi[ms].q[0] = *(const uint4*)&sA[buf][0][m][kbA];
            aHi[ms].q[1] = *(const uint4*)&sA[buf][0][m][kbA + 16];
            aLo[ms].q[0] = *(const uint4*)&sA[buf][1][m][kbA];
            aLo[ms].q[1] = *(const uint4*)&sA[buf][1][m][kbA + 16];
        }
#pragma unroll
        for (int ns = 0; ns < 2; ++ns) {
            int n = wn * 32 + ns * 16 + lh;
            bHi[ns].q[0] = *(const uint4*)&sB[buf][0][n][kbB];
            bHi[ns].q[1] = *(const uint4*)&sB[buf][0][n][kbB + 16];
            bLo[ns].q[0] = *(const uint4*)&sB[buf][1][n][kbB];
            bLo[ns].q[1] = *(const uint4*)&sB[buf][1][n][kbB + 16];
        }

#pragma unroll
        for (int ms = 0; ms < 2; ++ms)
#pragma unroll
            for (int ns = 0; ns < 2; ++ns) {
                acc[ms][ns] = __builtin_amdgcn_wmma_f32_16x16x32_bf16(
                    false, aHi[ms].v, false, bHi[ns].v, (short)0, acc[ms][ns], false, false);
                acc[ms][ns] = __builtin_amdgcn_wmma_f32_16x16x32_bf16(
                    false, aHi[ms].v, false, bLo[ns].v, (short)0, acc[ms][ns], false, false);
                acc[ms][ns] = __builtin_amdgcn_wmma_f32_16x16x32_bf16(
                    false, aLo[ms].v, false, bHi[ns].v, (short)0, acc[ms][ns], false, false);
            }
        __syncthreads();
    }

    // ---- epilogue: bias + relu, then store or pool-accumulate ----
#pragma unroll
    for (int ms = 0; ms < 2; ++ms)
#pragma unroll
        for (int ns = 0; ns < 2; ++ns) {
            int col   = nBase + wn * 32 + ns * 16 + lh;
            float bv  = bias[col];
            int rbase = mBase + wm * 32 + ms * 16 + (hiHalf ? 8 : 0);
            if (!poolMode) {
#pragma unroll
                for (int r = 0; r < 8; ++r) {
                    int row = rbase + r;
                    if (row < M) {
                        float v = acc[ms][ns][r] + bv;
                        v = v > 0.0f ? v : 0.0f;
                        Cout[(size_t)row * N + col] = v;
                    }
                }
            } else if (fastGraph >= 0) {
                float s = 0.0f;
#pragma unroll
                for (int r = 0; r < 8; ++r) {
                    float v = acc[ms][ns][r] + bv;
                    s += (v > 0.0f ? v : 0.0f);
                }
                s += __shfl_xor(s, 16, 32);
                if (!hiHalf)
                    atomicAdd(&poolSums[(size_t)fastGraph * N + col], s);
            } else {
#pragma unroll
                for (int r = 0; r < 8; ++r) {
                    int row = rbase + r;
                    if (row < M) {
                        float v = acc[ms][ns][r] + bv;
                        v = v > 0.0f ? v : 0.0f;
                        atomicAdd(&poolSums[(size_t)batch[row] * N + col], v);
                    }
                }
            }
        }
}

// ---------------- pre-pass: build bf16 hi/lo planes ----------------
// A-cat planes: [M,K] where col k<ksplit -> Agg[row,k]/max(deg,1), else Self[row,k-ksplit]
__global__ void split_cat_kernel(const float* __restrict__ Agg, const float* __restrict__ Self,
                                 const float* __restrict__ deg,
                                 __bf16* __restrict__ hi, __bf16* __restrict__ lo,
                                 int ksplit, int K, long total)
{
    long i = (long)blockIdx.x * blockDim.x + threadIdx.x;
    if (i >= total) return;
    int row = (int)(i / K);
    int k   = (int)(i - (long)row * K);
    float v;
    if (k < ksplit) v = Agg[(size_t)row * ksplit + k] / fmaxf(deg[row], 1.0f);
    else            v = Self[(size_t)row * (K - ksplit) + (k - ksplit)];
    __bf16 h = (__bf16)v;
    hi[i] = h;
    lo[i] = (__bf16)(v - (float)h);
}

// B^T planes: out[n*K+k] from B0[k,n] (k<ksplit) else B1[k-ksplit,n]
__global__ void splitT_kernel(const float* __restrict__ B0, const float* __restrict__ B1,
                              __bf16* __restrict__ hiT, __bf16* __restrict__ loT,
                              int ksplit, int K, int N)
{
    long i = (long)blockIdx.x * blockDim.x + threadIdx.x;
    if (i >= (long)K * N) return;
    int n = (int)(i / K);
    int k = (int)(i - (long)n * K);
    float v = (k < ksplit) ? B0[(size_t)k * N + n]
                           : B1[(size_t)(k - ksplit) * N + n];
    __bf16 h = (__bf16)v;
    hiT[i] = h;
    loT[i] = (__bf16)(v - (float)h);
}

// ---------------- support kernels ----------------
__global__ void deg_kernel(const int* __restrict__ dst, float* __restrict__ deg, int nE)
{
    int i = blockIdx.x * blockDim.x + threadIdx.x;
    if (i < nE) atomicAdd(&deg[dst[i]], 1.0f);
}

__global__ void gcnt_kernel(const int* __restrict__ batch, float* __restrict__ g, int n)
{
    int i = blockIdx.x * blockDim.x + threadIdx.x;
    if (i < n) atomicAdd(&g[batch[i]], 1.0f);
}

// one wave per edge: accum[dst] += feat[src]
__global__ void scatter_kernel(const int* __restrict__ src, const int* __restrict__ dst,
                               const float* __restrict__ feat, float* __restrict__ accum,
                               int D, int nE)
{
    int wave = (blockIdx.x * blockDim.x + threadIdx.x) >> 5;
    int lane = threadIdx.x & 31;
    if (wave >= nE) return;
    int s = src[wave], d = dst[wave];
    const float* f = feat + (size_t)s * D;
    float* a = accum + (size_t)d * D;
    for (int i = lane; i < D; i += 32)
        atomicAdd(&a[i], f[i]);
}

__global__ void pooldiv_kernel(const float* __restrict__ sums, const float* __restrict__ gcnt,
                               float* __restrict__ mean, int N)
{
    int i = blockIdx.x * blockDim.x + threadIdx.x;
    if (i >= N_GRAPHS * N) return;
    int g = i / N;
    mean[i] = sums[i] / fmaxf(gcnt[g], 1.0f);
}

// out[g,c] = b[c] + sum_k in[g,k] * (dup ? (W[k,c]+W[k+K,c]) : W[k,c])   (M=8, fp32 VALU)
__global__ void head_kernel(const float* __restrict__ in, const float* __restrict__ W,
                            const float* __restrict__ b, float* __restrict__ out,
                            int K, int N, int dupK)
{
    int c = blockIdx.x * blockDim.x + threadIdx.x;
    int g = blockIdx.y;
    if (c >= N) return;
    const float* row = in + (size_t)g * K;
    float s = b[c];
    if (!dupK) {
        for (int k = 0; k < K; ++k)
            s = fmaf(row[k], W[(size_t)k * N + c], s);
    } else {
        for (int k = 0; k < K; ++k)
            s = fmaf(row[k], W[(size_t)k * N + c] + W[(size_t)(k + K) * N + c], s);
    }
    out[(size_t)g * N + c] = s;
}

// ---------------- launcher ----------------
extern "C" void kernel_launch(void* const* d_in, const int* in_sizes, int n_in,
                              void* d_out, int out_size, void* d_ws, size_t ws_size,
                              hipStream_t stream)
{
    (void)in_sizes; (void)n_in; (void)out_size; (void)ws_size;

    const float* x    = (const float*)d_in[0];
    const int*   ei   = (const int*)d_in[1];   // JAX x64 off -> int32; [2, N_EDGES]
    const int*   bat  = (const int*)d_in[2];
    const float* W1n  = (const float*)d_in[3];
    const float* W1r  = (const float*)d_in[4];
    const float* b1   = (const float*)d_in[5];
    const float* W2n  = (const float*)d_in[6];
    const float* W2r  = (const float*)d_in[7];
    const float* b2   = (const float*)d_in[8];
    const float* Wp   = (const float*)d_in[9];
    const float* bp   = (const float*)d_in[10];
    const float* Wg   = (const float*)d_in[11];
    const float* bg   = (const float*)d_in[12];
    float* out = (float*)d_out;

    const int* srcIdx = ei;
    const int* dstIdx = ei + N_EDGES;

    char* ws = (char*)d_ws;
    size_t off = 0;
    auto take = [&](size_t bytes) -> char* {
        char* p = ws + off;
        off = (off + bytes + 255) & ~(size_t)255;
        return p;
    };
    float*  deg        = (float*)take((size_t)N_NODES * 4);
    float*  gcnt       = (float*)take((size_t)N_GRAPHS * 4);
    float*  poolSums   = (float*)take((size_t)N_GRAPHS * D_OUT * 4);
    float*  pooledMean = (float*)take((size_t)N_GRAPHS * D_OUT * 4);
    float*  aggBuf     = (float*)take((size_t)N_NODES * D_HID * 4);          // agg1 then agg2
    float*  h1         = (float*)take((size_t)N_NODES * D_HID * 4);
    __bf16* planes     = (__bf16*)take((size_t)N_NODES * 1024 * 2 * 2);      // A hi+lo, reused
    __bf16* bt1        = (__bf16*)take((size_t)768 * 512 * 2 * 2);           // B1^T hi+lo
    __bf16* bt2        = (__bf16*)take((size_t)1024 * 2560 * 2 * 2);         // B2^T hi+lo

    const int K1 = D_IN + D_IN;    // 768
    const int K2 = D_HID + D_HID;  // 1024
    __bf16* a1hi = planes;                         __bf16* a1lo = planes + (size_t)N_NODES * K1;
    __bf16* a2hi = planes;                         __bf16* a2lo = planes + (size_t)N_NODES * K2;
    __bf16* bt1hi = bt1;                           __bf16* bt1lo = bt1 + (size_t)K1 * D_HID;
    __bf16* bt2hi = bt2;                           __bf16* bt2lo = bt2 + (size_t)K2 * D_OUT;

    hipMemsetAsync(deg, 0, (size_t)N_NODES * 4, stream);
    hipMemsetAsync(gcnt, 0, (size_t)N_GRAPHS * 4, stream);
    hipMemsetAsync(poolSums, 0, (size_t)N_GRAPHS * D_OUT * 4, stream);
    hipMemsetAsync(aggBuf, 0, (size_t)N_NODES * D_IN * 4, stream);

    deg_kernel<<<(N_EDGES + 255) / 256, 256, 0, stream>>>(dstIdx, deg, N_EDGES);
    gcnt_kernel<<<(N_NODES + 255) / 256, 256, 0, stream>>>(bat, gcnt, N_NODES);

    // weights -> transposed bf16 hi/lo planes (done inside the capture; cheap)
    splitT_kernel<<<(unsigned)(((long)K1 * D_HID + 255) / 256), 256, 0, stream>>>(
        W1n, W1r, bt1hi, bt1lo, D_IN, K1, D_HID);
    splitT_kernel<<<(unsigned)(((long)K2 * D_OUT + 255) / 256), 256, 0, stream>>>(
        W2n, W2r, bt2hi, bt2lo, D_HID, K2, D_OUT);

    // ---- Layer 1 ----
    scatter_kernel<<<(N_EDGES * 32 + 255) / 256, 256, 0, stream>>>(
        srcIdx, dstIdx, x, aggBuf, D_IN, N_EDGES);
    {
        long total = (long)N_NODES * K1;
        split_cat_kernel<<<(unsigned)((total + 255) / 256), 256, 0, stream>>>(
            aggBuf, x, deg, a1hi, a1lo, D_IN, K1, total);
    }
    {
        dim3 grid(D_HID / BN, (N_NODES + BM - 1) / BM);
        wmma_gemm_bf16<<<grid, 256, 0, stream>>>(a1hi, a1lo, bt1hi, bt1lo, b1,
                                                 h1, nullptr, bat,
                                                 N_NODES, D_HID, K1, 0);
    }

    // ---- Layer 2 (fused with global mean pool numerator) ----
    hipMemsetAsync(aggBuf, 0, (size_t)N_NODES * D_HID * 4, stream);
    scatter_kernel<<<(N_EDGES * 32 + 255) / 256, 256, 0, stream>>>(
        srcIdx, dstIdx, h1, aggBuf, D_HID, N_EDGES);
    {
        long total = (long)N_NODES * K2;
        split_cat_kernel<<<(unsigned)((total + 255) / 256), 256, 0, stream>>>(
            aggBuf, h1, deg, a2hi, a2lo, D_HID, K2, total);
    }
    {
        dim3 grid(D_OUT / BN, (N_NODES + BM - 1) / BM);
        wmma_gemm_bf16<<<grid, 256, 0, stream>>>(a2hi, a2lo, bt2hi, bt2lo, b2,
                                                 nullptr, poolSums, bat,
                                                 N_NODES, D_OUT, K2, 1);
    }

    pooldiv_kernel<<<(N_GRAPHS * D_OUT + 255) / 256, 256, 0, stream>>>(
        poolSums, gcnt, pooledMean, D_OUT);

    dim3 gh((D_OUT + 255) / 256, N_GRAPHS);
    head_kernel<<<gh, 256, 0, stream>>>(pooledMean, Wp, bp, out, D_OUT, D_OUT, 0);
    head_kernel<<<gh, 256, 0, stream>>>(out, Wg, bg, out + (size_t)N_GRAPHS * D_OUT,
                                        D_OUT, D_OUT, 1);
}
